// AdvancedCrossAttentionMerger_1726576853167
// MI455X (gfx1250) — matrix operand order repaired and checked by reading, not compiled, not run on hardware
//
#include <hip/hip_runtime.h>
#include <hip/hip_bf16.h>
#include <math.h>
#include <stdint.h>

// ---------------------------------------------------------------------------
// MHA: Y = softmax((XqWq^T+bq)(XkWk^T+bk)^T/sqrt(HD)) (XvWv^T+bv) Wo^T + bo
// B=2, SQ=SKV=2048, D=1024, H=16, HD=64.
//
// Pipeline:
//   1) bulk fp32 -> bf16 conversion of activations + weights
//   2) Q/K projections: TDM-staged bf16 GEMM (V_WMMA_F32_16X16X32_BF16)
//      V projection: same GEMM, epilogue writes transposed Vt[b,h,hd,skv]
//   3) flash attention: K/Vt tiles staged via TDM (double-buffered),
//      bf16 P staged in LDS, f32 softmax/accumulate
//   4) output projection GEMM -> fp32 d_out
// ---------------------------------------------------------------------------

#define B_   2
#define SQ_  2048
#define SKV_ 2048
#define D_   1024
#define H_   16
#define HD_  64

typedef uint32_t u32;
typedef __attribute__((ext_vector_type(16))) __bf16 v16bf;
typedef __attribute__((ext_vector_type(8)))  float  v8f;
typedef __attribute__((ext_vector_type(4)))  u32    u32x4;
typedef __attribute__((ext_vector_type(8)))  u32    u32x8;

union AFrag { v16bf v; __bf16 e[16]; uint4 q[2]; };
union CFrag { v8f   v; float  e[8]; };
union PackBF2 { u32 u; __bf16 h[2]; };

__device__ __forceinline__ v8f wmma_bf16(v16bf a, v16bf b, v8f c) {
  return __builtin_amdgcn_wmma_f32_16x16x32_bf16(
      false, a, false, b, (short)0, c, false, false);
}

// ---------------------------------------------------------------------------
// Fragment loaders (ISA 7.12.2 wave32 layouts). All sources 16B-aligned rows.
// ---------------------------------------------------------------------------

// A fragment 16x32 (MxK) from row-major bf16 [*, ld]:
// lane<16: M=lane, K=0..7 & 16..23 ; lane>=16: M=lane-16, K=8..15 & 24..31.
__device__ __forceinline__ v16bf load_A_frag_bf16(const __bf16* base, int ld, int lane) {
  const int m     = lane & 15;
  const int khalf = (lane < 16) ? 0 : 8;
  const __bf16* p = base + (size_t)m * ld + khalf;
  AFrag a;
  a.q[0] = *(const uint4*)p;          // K = khalf .. khalf+7
  a.q[1] = *(const uint4*)(p + 16);   // K = 16+khalf .. 16+khalf+7
  return a.v;
}

// B fragment 32x16 (KxN) with B[k][n] = base[n*ld + k] (source row-major [N][K]):
// lane: n = lane&15; lanes<16 hold K=0..7 & 16..23, lanes>=16 hold K=8..15 & 24..31.
__device__ __forceinline__ v16bf load_B_fragT_bf16(const __bf16* base, int ld, int lane) {
  const int n  = lane & 15;
  const int kh = (lane < 16) ? 0 : 8;
  const __bf16* p = base + (size_t)n * ld + kh;
  AFrag b;
  b.q[0] = *(const uint4*)p;          // K = kh .. kh+7
  b.q[1] = *(const uint4*)(p + 16);   // K = 16+kh .. 16+kh+7
  return b.v;
}

// ---------------------------------------------------------------------------
// Tensor Data Mover: async 2D bf16 tile load global -> LDS (ISA ch. 8).
// D# group0 (4 SGPRs) + group1 (8 SGPRs); 2-group form => 2D tensors.
// Tile lands in LDS row-major [tile_d1][tile_d0], contiguous.
// ---------------------------------------------------------------------------
__device__ __forceinline__ void tdm_load_2d_bf16(
    u32 lds_byte_addr, const __bf16* gptr,
    u32 tile_d0, u32 tile_d1,   // tile extents in elements
    u32 d0_stride)              // source row stride in elements
{
  const uint64_t ga = (uint64_t)(uintptr_t)gptr;
  u32x4 g0;
  g0[0] = 1u;                                          // count=1, user mode
  g0[1] = lds_byte_addr;                               // lds_addr
  g0[2] = (u32)ga;                                     // global_addr[31:0]
  g0[3] = (u32)((ga >> 32) & 0x1FFFFFFu) | (2u << 30); // addr[56:32] | type=2
  u32x8 g1;
  g1[0] = (1u << 16);                                  // data_size=1 (2B)
  g1[1] = (tile_d0 & 0xFFFFu) << 16;                   // tensor_dim0 = tile_d0
  g1[2] = (tile_d0 >> 16) | ((tile_d1 & 0xFFFFu) << 16);  // | tensor_dim1 lo
  g1[3] = (tile_d1 >> 16) | (tile_d0 << 16);           // | tile_dim0
  g1[4] = tile_d1 & 0xFFFFu;                           // tile_dim1 (tile_dim2=0)
  g1[5] = d0_stride;                                   // tensor_dim0_stride[31:0]
  g1[6] = 0;
  g1[7] = 0;
  asm volatile("tensor_load_to_lds %0, %1" :: "s"(g0), "s"(g1) : "memory");
}

// ---------------------------------------------------------------------------
// GEMM: Y[M,N] = Xbf[M,K] @ Wbf[N,K]^T + bias[N]
// Block 256 threads (8 waves): block tile 128x64, wave tile 32x32, k-step 32.
// A/B tiles staged in LDS by wave 0 via TDM, double-buffered.
// TRANSV: write output as Vt[(b*D + c), s] with r = b*SKV + s (for V proj).
// ---------------------------------------------------------------------------
template <typename OutT, bool TRANSV>
__global__ __launch_bounds__(256) void gemm_bias_wmma(
    const __bf16* __restrict__ X, const __bf16* __restrict__ W,
    const float* __restrict__ bias, OutT* __restrict__ Y,
    int M, int N, int K) {
  __shared__ __align__(16) __bf16 sA[2][128 * 32];  // [buf][row][k]
  __shared__ __align__(16) __bf16 sB[2][64 * 32];   // [buf][col][k]

  const int lane = threadIdx.x & 31;
  const int wave = threadIdx.x >> 5;   // 0..7
  const int wm = wave >> 1;            // 0..3 -> rows wm*32
  const int wn = wave & 1;             // 0..1 -> cols wn*32
  const int row0 = blockIdx.x * 128;
  const int col0 = blockIdx.y * 64;

  v8f acc[2][2];
#pragma unroll
  for (int i = 0; i < 2; ++i)
#pragma unroll
    for (int j = 0; j < 2; ++j) acc[i][j] = (v8f)(0.0f);

  const int KT = K / 32;
  if (wave == 0) {
    tdm_load_2d_bf16((u32)(uintptr_t)&sA[0][0], X + (size_t)row0 * K, 32u, 128u, (u32)K);
    tdm_load_2d_bf16((u32)(uintptr_t)&sB[0][0], W + (size_t)col0 * K, 32u, 64u, (u32)K);
    __builtin_amdgcn_s_wait_tensorcnt(0);
  }
  __syncthreads();

  for (int kt = 0; kt < KT; ++kt) {
    const int cur = kt & 1;
    const int nxt = cur ^ 1;
    if (wave == 0 && (kt + 1) < KT) {
      const int kk = (kt + 1) * 32;
      tdm_load_2d_bf16((u32)(uintptr_t)&sA[nxt][0], X + (size_t)row0 * K + kk, 32u, 128u, (u32)K);
      tdm_load_2d_bf16((u32)(uintptr_t)&sB[nxt][0], W + (size_t)col0 * K + kk, 32u, 64u, (u32)K);
    }

    // preload all four fragments, then issue all WMMAs back-to-back
    const v16bf a0 = load_A_frag_bf16(&sA[cur][(wm * 32) * 32], 32, lane);
    const v16bf a1 = load_A_frag_bf16(&sA[cur][(wm * 32 + 16) * 32], 32, lane);
    const v16bf b0 = load_B_fragT_bf16(&sB[cur][(wn * 32) * 32], 32, lane);
    const v16bf b1 = load_B_fragT_bf16(&sB[cur][(wn * 32 + 16) * 32], 32, lane);
    acc[0][0] = wmma_bf16(a0, b0, acc[0][0]);
    acc[0][1] = wmma_bf16(a0, b1, acc[0][1]);
    acc[1][0] = wmma_bf16(a1, b0, acc[1][0]);
    acc[1][1] = wmma_bf16(a1, b1, acc[1][1]);

    if (wave == 0) __builtin_amdgcn_s_wait_tensorcnt(0);
    __syncthreads();
  }

  // C layout: lane -> col (lane&15), VGPR j -> row j + (lane<16 ? 0 : 8).
  const int n = lane & 15;
  const int rbase = (lane < 16) ? 0 : 8;
#pragma unroll
  for (int ti = 0; ti < 2; ++ti) {
#pragma unroll
    for (int tj = 0; tj < 2; ++tj) {
      const int c = col0 + wn * 32 + tj * 16 + n;
      const float bv = bias[c];
      CFrag cf; cf.v = acc[ti][tj];
#pragma unroll
      for (int j = 0; j < 8; ++j) {
        const int r = row0 + wm * 32 + ti * 16 + rbase + j;
        const float val = cf.e[j] + bv;
        if constexpr (TRANSV) {
          // r = b*SKV + s ; Vt index = (b*D + c)*SKV + s
          const int bb = r / SKV_;
          const int s  = r - bb * SKV_;
          Y[((size_t)bb * D_ + c) * SKV_ + s] = (OutT)val;
        } else {
          Y[(size_t)r * N + c] = (OutT)val;
        }
      }
    }
  }
}

// ---------------------------------------------------------------------------
// Flash attention. Q/K stored bf16 [B, S, D]; V stored transposed
// Vt[b, h, hd, skv]. Grid: (SQ/64, H, B). Block 128 = 4 waves x 16 q rows.
// K/V 64x64 tiles staged in LDS via TDM, double-buffered.
// ---------------------------------------------------------------------------
#define SROW 68  // padded LDS row stride (floats; row = 17 float4s)

__global__ __launch_bounds__(128) void attn_wmma(
    const __bf16* __restrict__ Qp, const __bf16* __restrict__ Kp,
    const __bf16* __restrict__ Vt, __bf16* __restrict__ Ctx) {
  __shared__ __align__(16) __bf16 sK[2][64 * 64];  // [buf][kv][hd]
  __shared__ __align__(16) __bf16 sV[2][64 * 64];  // [buf][hd][kv]
  __shared__ __align__(16) float  sS[4][16][SROW];
  __shared__ __align__(16) __bf16 sP[4][16][64];
  __shared__ float sBcast[4][16];

  const int lane = threadIdx.x & 31;
  const int wave = threadIdx.x >> 5;  // 0..3
  const int h = blockIdx.y;
  const int b = blockIdx.z;
  const int qrow0 = blockIdx.x * 64 + wave * 16;

  const __bf16* Kbase  = Kp + (size_t)b * SKV_ * D_ + h * HD_;    // [kv][D]
  const __bf16* Vtbase = Vt + ((size_t)b * D_ + h * HD_) * SKV_;  // [hd][SKV]

  const size_t qoff = ((size_t)b * SQ_ + qrow0) * D_ + h * HD_;
  const v16bf qa0 = load_A_frag_bf16(Qp + qoff, D_, lane);       // K 0..31
  const v16bf qa1 = load_A_frag_bf16(Qp + qoff + 32, D_, lane);  // K 32..63

  v8f acc[4];
#pragma unroll
  for (int j = 0; j < 4; ++j) acc[j] = (v8f)(0.0f);

  float rm = -INFINITY;  // running row max (lanes 0..15)
  float rl = 0.0f;       // running row sum

  const int n = lane & 15;
  const int rbase = (lane < 16) ? 0 : 8;
  const int NT = SKV_ / 64;

  if (wave == 0) {
    tdm_load_2d_bf16((u32)(uintptr_t)&sK[0][0], Kbase, 64u, 64u, (u32)D_);
    tdm_load_2d_bf16((u32)(uintptr_t)&sV[0][0], Vtbase, 64u, 64u, (u32)SKV_);
    __builtin_amdgcn_s_wait_tensorcnt(0);
  }
  __syncthreads();

  for (int kt = 0; kt < NT; ++kt) {
    const int cur = kt & 1;
    const int nxt = cur ^ 1;
    if (wave == 0 && (kt + 1) < NT) {
      const int kv1 = (kt + 1) * 64;
      tdm_load_2d_bf16((u32)(uintptr_t)&sK[nxt][0], Kbase + (size_t)kv1 * D_, 64u, 64u, (u32)D_);
      tdm_load_2d_bf16((u32)(uintptr_t)&sV[nxt][0], Vtbase + kv1, 64u, 64u, (u32)SKV_);
    }

    // ---- S tile: preload all 8 K fragments, then 8 WMMAs ----
    v16bf kb[4][2];
#pragma unroll
    for (int jt = 0; jt < 4; ++jt) {
      const __bf16* kbp = &sK[cur][(jt * 16) * 64];  // B[k][n] = sK[n][k], ld=64
      kb[jt][0] = load_B_fragT_bf16(kbp, 64, lane);       // k 0..31
      kb[jt][1] = load_B_fragT_bf16(kbp + 32, 64, lane);  // k 32..63
    }
#pragma unroll
    for (int jt = 0; jt < 4; ++jt) {
      v8f s = wmma_bf16(qa0, kb[jt][0], (v8f)(0.0f));
      s = wmma_bf16(qa1, kb[jt][1], s);
      CFrag c; c.v = s;
#pragma unroll
      for (int j = 0; j < 8; ++j)
        sS[wave][rbase + j][jt * 16 + n] = c.e[j] * 0.125f;
    }
    __syncthreads();

    // ---- online softmax: lane r (<16) owns q-row r; writes packed bf16 P ----
    if (lane < 16) {
      const float4* row = (const float4*)&sS[wave][lane][0];
      float tmax = -INFINITY;
#pragma unroll
      for (int c = 0; c < 16; ++c) {
        const float4 v = row[c];
        tmax = fmaxf(tmax, fmaxf(fmaxf(v.x, v.y), fmaxf(v.z, v.w)));
      }
      const float mnew = fmaxf(rm, tmax);
      const float alpha = __expf(rm - mnew);
      rm = mnew;
      float psum = 0.0f;
      u32* prow = (u32*)&sP[wave][lane][0];
#pragma unroll
      for (int c = 0; c < 16; ++c) {
        const float4 v = row[c];
        const float p0 = __expf(v.x - mnew);
        const float p1 = __expf(v.y - mnew);
        const float p2 = __expf(v.z - mnew);
        const float p3 = __expf(v.w - mnew);
        psum += (p0 + p1) + (p2 + p3);
        PackBF2 q0, q1;
        q0.h[0] = (__bf16)p0; q0.h[1] = (__bf16)p1;
        q1.h[0] = (__bf16)p2; q1.h[1] = (__bf16)p3;
        prow[2 * c]     = q0.u;
        prow[2 * c + 1] = q1.u;
      }
      rl = rl * alpha + psum;
      sBcast[wave][lane] = alpha;
    }
    __syncthreads();

    // rescale accumulators by per-row alpha
#pragma unroll
    for (int jt = 0; jt < 4; ++jt) {
      CFrag c; c.v = acc[jt];
#pragma unroll
      for (int j = 0; j < 8; ++j) c.e[j] *= sBcast[wave][rbase + j];
      acc[jt] = c.v;
    }

    // ---- ctx += P @ V : preload P + all 8 V fragments, then 8 WMMAs ----
    const v16bf pa0 = load_A_frag_bf16(&sP[wave][0][0], 64, lane);       // kv 0..31
    const v16bf pa1 = load_A_frag_bf16(&sP[wave][0][0] + 32, 64, lane);  // kv 32..63
    v16bf vb[4][2];
#pragma unroll
    for (int jt = 0; jt < 4; ++jt) {
      const __bf16* vbp = &sV[cur][(jt * 16) * 64];  // B[k][n] = sV[n][k], ld=64
      vb[jt][0] = load_B_fragT_bf16(vbp, 64, lane);       // kv 0..31
      vb[jt][1] = load_B_fragT_bf16(vbp + 32, 64, lane);  // kv 32..63
    }
#pragma unroll
    for (int jt = 0; jt < 4; ++jt) {
      acc[jt] = wmma_bf16(pa0, vb[jt][0], acc[jt]);
      acc[jt] = wmma_bf16(pa1, vb[jt][1], acc[jt]);
    }

    if (wave == 0) __builtin_amdgcn_s_wait_tensorcnt(0);
    __syncthreads();  // next iteration overwrites sS/sP and swaps buffers
  }

  if (lane < 16) sBcast[wave][lane] = 1.0f / rl;
  __syncthreads();

#pragma unroll
  for (int jt = 0; jt < 4; ++jt) {
    CFrag c; c.v = acc[jt];
#pragma unroll
    for (int j = 0; j < 8; ++j) {
      const int r = qrow0 + rbase + j;
      Ctx[((size_t)b * SQ_ + r) * D_ + h * HD_ + jt * 16 + n] =
          (__bf16)(c.e[j] * sBcast[wave][rbase + j]);
    }
  }
}

// ---------------------------------------------------------------------------
// fp32 -> bf16 bulk conversion (n multiple of 4).
// ---------------------------------------------------------------------------
__global__ __launch_bounds__(256) void cvt_f32_bf16(
    const float* __restrict__ in, __bf16* __restrict__ out, int n) {
  const int i = (blockIdx.x * 256 + threadIdx.x) * 4;
  if (i < n) {
    const float4 v = *(const float4*)(in + i);
    out[i + 0] = (__bf16)v.x;
    out[i + 1] = (__bf16)v.y;
    out[i + 2] = (__bf16)v.z;
    out[i + 3] = (__bf16)v.w;
  }
}

// ---------------------------------------------------------------------------
// Launch.
// Workspace (bf16): Xq|Xk|Xv (3x4M) | Wq|Wk|Wv|Wo (4x1M) | Qp|Kp|Vt (3x4M)
//                 | Ctx (4M)  => ~67 MB.
// ---------------------------------------------------------------------------
extern "C" void kernel_launch(void* const* d_in, const int* in_sizes, int n_in,
                              void* d_out, int out_size, void* d_ws, size_t ws_size,
                              hipStream_t stream) {
  const float* query = (const float*)d_in[0];
  const float* key   = (const float*)d_in[1];
  const float* value = (const float*)d_in[2];
  const float* Wq = (const float*)d_in[3];
  const float* bq = (const float*)d_in[4];
  const float* Wk = (const float*)d_in[5];
  const float* bk = (const float*)d_in[6];
  const float* Wv = (const float*)d_in[7];
  const float* bv = (const float*)d_in[8];
  const float* Wo = (const float*)d_in[9];
  const float* bo = (const float*)d_in[10];

  const size_t SD = (size_t)B_ * SQ_ * D_;  // 4M elements
  const size_t DD = (size_t)D_ * D_;        // 1M elements
  __bf16* p = (__bf16*)d_ws;
  __bf16 *Xq = p, *Xk = Xq + SD, *Xv = Xk + SD;
  __bf16 *Wqb = Xv + SD, *Wkb = Wqb + DD, *Wvb = Wkb + DD, *Wob = Wvb + DD;
  __bf16 *Qp = Wob + DD, *Kp = Qp + SD, *Vt = Kp + SD, *Ctx = Vt + SD;

  // 1) bulk fp32 -> bf16
  const int cb = 256;
  cvt_f32_bf16<<<dim3((u32)((SD / 4 + cb - 1) / cb)), cb, 0, stream>>>(query, Xq, (int)SD);
  cvt_f32_bf16<<<dim3((u32)((SD / 4 + cb - 1) / cb)), cb, 0, stream>>>(key,   Xk, (int)SD);
  cvt_f32_bf16<<<dim3((u32)((SD / 4 + cb - 1) / cb)), cb, 0, stream>>>(value, Xv, (int)SD);
  cvt_f32_bf16<<<dim3((u32)((DD / 4 + cb - 1) / cb)), cb, 0, stream>>>(Wq, Wqb, (int)DD);
  cvt_f32_bf16<<<dim3((u32)((DD / 4 + cb - 1) / cb)), cb, 0, stream>>>(Wk, Wkb, (int)DD);
  cvt_f32_bf16<<<dim3((u32)((DD / 4 + cb - 1) / cb)), cb, 0, stream>>>(Wv, Wvb, (int)DD);
  cvt_f32_bf16<<<dim3((u32)((DD / 4 + cb - 1) / cb)), cb, 0, stream>>>(Wo, Wob, (int)DD);

  // 2) projections (V written transposed)
  const int M = B_ * SQ_;            // 4096
  dim3 gg(M / 128, D_ / 64);         // 32 x 16
  gemm_bias_wmma<__bf16, false><<<gg, 256, 0, stream>>>(Xq, Wqb, bq, Qp, M, D_, D_);
  gemm_bias_wmma<__bf16, false><<<gg, 256, 0, stream>>>(Xk, Wkb, bk, Kp, M, D_, D_);
  gemm_bias_wmma<__bf16, true ><<<gg, 256, 0, stream>>>(Xv, Wvb, bv, Vt, M, D_, D_);

  // 3) attention
  attn_wmma<<<dim3(SQ_ / 64, H_, B_), 128, 0, stream>>>(Qp, Kp, Vt, Ctx);

  // 4) output projection -> fp32
  gemm_bias_wmma<float, false><<<gg, 256, 0, stream>>>(Ctx, Wob, bo, (float*)d_out, M, D_, D_);
}
